// ComformerEquivariant_45810121179808
// MI455X (gfx1250) — matrix operand order repaired and checked by reading, not compiled, not run on hardware
//
#include <hip/hip_runtime.h>
#include <math.h>

typedef __bf16 bf16_t;
typedef __attribute__((ext_vector_type(16))) __bf16 v16bf;
typedef __attribute__((ext_vector_type(8)))  float  v8f;

#define NN 12800
#define NE 102400
#define NG 512

// ---------------- device helpers ----------------

static __device__ __forceinline__ float sigmoid_f(float x){ return 1.f/(1.f+__expf(-x)); }
static __device__ __forceinline__ float silu_f(float x){ return x*sigmoid_f(x); }
static __device__ __forceinline__ float softplus_f(float x){ return x>20.f ? x : log1pf(__expf(x)); }

static __device__ __forceinline__ v8f zero8(){ v8f z={0.f,0.f,0.f,0.f,0.f,0.f,0.f,0.f}; return z; }

static __device__ __forceinline__ v8f wmma_bf16(v16bf a, v16bf b, v8f c){
  return __builtin_amdgcn_wmma_f32_16x16x32_bf16(false, a, false, b, (short)0, c, false, false);
}

// A fragment (16x32 bf16, ISA 7.12.2 layout) from an f32 row, converting in-register.
// lane L and L+16 share row M=(L&15); L<16 holds K {0..7,16..23}, L>=16 holds {8..15,24..31}.
// NO tail guard: every A source is padded to a multiple of 32 columns.
static __device__ __forceinline__ v16bf a_frag_f32(const float* __restrict__ arow, int k0){
  const int half = ((int)threadIdx.x >> 4) & 1;
  const float* p = arow + k0 + half*8;
  v16bf a;
#pragma unroll
  for (int i=0;i<8;++i){ a[i]=(bf16_t)p[i]; a[i+8]=(bf16_t)p[16+i]; }
  return a;
}

static __device__ __forceinline__ v16bf a_frag_bf16(const bf16_t* __restrict__ arow, int k0){
  const int half = ((int)threadIdx.x >> 4) & 1;
  const bf16_t* p = arow + k0 + half*8;
  v16bf a;
#pragma unroll
  for (int i=0;i<8;++i){ a[i]=p[i]; a[i+8]=p[16+i]; }
  return a;
}

// B fragment (32x16 bf16): weights stored transposed Wt[N][Kpad]; lane n = lane&15 reads
// K {0..7,16..23} (lane<16) or {8..15,24..31} (lane>=16), matching the A fragment.
static __device__ __forceinline__ v16bf b_frag(const bf16_t* __restrict__ Wt, int ldk, int n0, int k0){
  const int lane = (int)threadIdx.x & 31;
  const bf16_t* p = Wt + (size_t)(n0 + (lane & 15)) * ldk + k0 + ((lane >> 4) << 3);
  v16bf b;
#pragma unroll
  for (int i=0;i<8;++i){ b[i]=p[i]; b[i+8]=p[16+i]; }
  return b;
}

// ---------------- utility kernels ----------------

__global__ void zero_f32_kernel(float* p, long long n){
  long long i = (long long)blockIdx.x*blockDim.x + threadIdx.x;
  long long s = (long long)gridDim.x*blockDim.x;
  for (; i<n; i+=s) p[i]=0.f;
}

// W f32 [K][N] row-major  ->  Wt bf16 [N][Kpad], zero-padded in K.
__global__ void w_to_bf16T_kernel(const float* __restrict__ W, bf16_t* __restrict__ Wt,
                                  int K, int N, int Kpad){
  int i = blockIdx.x*blockDim.x + threadIdx.x;
  int tot = N*Kpad;
  if (i>=tot) return;
  int n = i / Kpad, k = i % Kpad;
  Wt[i] = (bf16_t)((k<K)? W[(size_t)k*N + n] : 0.f);
}

// pad x [NN][92] -> xp [NN][96] (zero tail) so atom-embed GEMM takes the fast path
__global__ void pad_x_kernel(const float* __restrict__ x, float* __restrict__ xp){
  long long i = (long long)blockIdx.x*blockDim.x + threadIdx.x;
  long long tot = (long long)NN*96;
  long long s = (long long)gridDim.x*blockDim.x;
  for (; i<tot; i+=s){
    int k = (int)(i % 96);
    int r = (int)(i / 96);
    xp[i] = (k < 92) ? x[(size_t)r*92 + k] : 0.f;
  }
}

// ---------------- generic WMMA GEMM: out[M][N] = act(A[M][Kpad] @ W + b) ----------------
// block = 256 threads (8 waves); blockIdx.x -> 16-row strip; wave -> 16-col tile.
__global__ void gemm_kernel(const float* __restrict__ A, int lda, int Kpad,
                            const bf16_t* __restrict__ Wt, const float* __restrict__ bias,
                            float* __restrict__ out, int N, int act, int accum){
  const int m0 = blockIdx.x*16;
  const int wave = (int)threadIdx.x>>5;
  const int lane = (int)threadIdx.x&31;
  const int n0 = ((int)blockIdx.y*8 + wave)*16;
  if (n0 >= N) return;
  const float* arow = A + (size_t)(m0 + (lane&15))*lda;
  v8f acc = zero8();
  if (Kpad == 128){
#pragma unroll
    for (int k0=0;k0<128;k0+=32)
      acc = wmma_bf16(a_frag_f32(arow,k0), b_frag(Wt,128,n0,k0), acc);
  } else {
    for (int k0=0;k0<Kpad;k0+=32)
      acc = wmma_bf16(a_frag_f32(arow,k0), b_frag(Wt,Kpad,n0,k0), acc);
  }
  const int n = n0 + (lane&15);
  const int mh = (lane>>4)*8;
  const float bv = bias ? bias[n] : 0.f;
#pragma unroll
  for (int j=0;j<8;++j){
    float v = acc[j] + bv;
    if (act==1) v = silu_f(v);
    else if (act==2) v = softplus_f(v);
    size_t o = (size_t)(m0 + mh + j)*N + n;
    if (accum) out[o] += v; else out[o] = v;
  }
}

// ---------------- RBF expansion + rbf_lin + softplus, fully fused ----------------
__global__ void rbf_ef_kernel(const float* __restrict__ ea, const bf16_t* __restrict__ wt_rbf,
                              const float* __restrict__ rbf_b, float* __restrict__ ef){
  __shared__ float dsh[16];
  __shared__ bf16_t rbfL[16][512];
  const int e0 = blockIdx.x*16;
  const int tid = (int)threadIdx.x;
  if (tid < 16){
    const float* p = ea + (size_t)(e0+tid)*3;
    float nrm = sqrtf(p[0]*p[0]+p[1]*p[1]+p[2]*p[2]);
    dsh[tid] = -0.75f / nrm;
  }
  __syncthreads();
  const float gamma = 511.0f/4.0f;
  for (int idx=tid; idx<16*512; idx+=blockDim.x){
    int r = idx>>9, k = idx&511;
    float c = -4.0f + (float)k*(4.0f/511.0f);
    float dd = dsh[r]-c;
    rbfL[r][k] = (bf16_t)__expf(-gamma*dd*dd);
  }
  __syncthreads();
  const int wave = tid>>5, lane = tid&31;
  const int n0 = wave*16;
  const bf16_t* arow = &rbfL[lane&15][0];
  v8f acc = zero8();
  for (int k0=0;k0<512;k0+=32)
    acc = wmma_bf16(a_frag_bf16(arow,k0), b_frag(wt_rbf,512,n0,k0), acc);
  const int n = n0 + (lane&15);
  const int mh = (lane>>4)*8;
  const float bv = rbf_b[n];
#pragma unroll
  for (int j=0;j<8;++j)
    ef[(size_t)(e0+mh+j)*128 + n] = softplus_f(acc[j]+bv);
}

// ---------------- fused per-edge attention path (edge-lin, key/msg MLPs, alpha) ----------------
__global__ void conv_edge_kernel(const float* __restrict__ ef,
    const float* __restrict__ qb, const float* __restrict__ kb, const float* __restrict__ vb,
    const int* __restrict__ src, const int* __restrict__ dst,
    const bf16_t* __restrict__ wte, const float* __restrict__ eb,
    const bf16_t* __restrict__ wtk1, const float* __restrict__ k1b,
    const bf16_t* __restrict__ wtk2, const float* __restrict__ k2b,
    const bf16_t* __restrict__ wtm1, const float* __restrict__ m1b,
    const bf16_t* __restrict__ wtm2, const float* __restrict__ m2b,
    float* __restrict__ alpha, float* __restrict__ msg){
  __shared__ float eeL[16][128];
  __shared__ float hL[16][128];
  __shared__ int sids[16], dids[16];
  const int e0 = blockIdx.x*16;
  const int tid = (int)threadIdx.x;
  const int wave = tid>>5, lane = tid&31;
  const int mrow = lane&15, mh = (lane>>4)*8;
  const int n0 = wave*16;
  const int n = n0 + mrow;
  if (tid < 16){ sids[tid] = src[e0+tid]; dids[tid] = dst[e0+tid]; }
  __syncthreads();
  { // ee = ef @ edgeW + eb
    const float* arow = ef + (size_t)(e0+mrow)*128;
    v8f acc = zero8();
#pragma unroll
    for (int k0=0;k0<128;k0+=32) acc = wmma_bf16(a_frag_f32(arow,k0), b_frag(wte,128,n0,k0), acc);
    const float bv = eb[n];
#pragma unroll
    for (int j=0;j<8;++j) eeL[mh+j][n] = acc[j]+bv;
  }
  __syncthreads();
  { // h = silu([k[dst], k[src], ee] @ key1 + b)   (K=384, straight-line per source)
    const float* ad = kb + (size_t)dids[mrow]*128;
    const float* as = kb + (size_t)sids[mrow]*128;
    const float* ae = &eeL[mrow][0];
    v8f acc = zero8();
#pragma unroll
    for (int c=0;c<4;++c) acc = wmma_bf16(a_frag_f32(ad,c*32), b_frag(wtk1,384,n0,c*32), acc);
#pragma unroll
    for (int c=0;c<4;++c) acc = wmma_bf16(a_frag_f32(as,c*32), b_frag(wtk1,384,n0,128+c*32), acc);
#pragma unroll
    for (int c=0;c<4;++c) acc = wmma_bf16(a_frag_f32(ae,c*32), b_frag(wtk1,384,n0,256+c*32), acc);
    const float bv = k1b[n];
#pragma unroll
    for (int j=0;j<8;++j) hL[mh+j][n] = silu_f(acc[j]+bv);
  }
  __syncthreads();
  { // kj = h @ key2 + b ; alpha = q[dst]*kj/sqrt(128)
    const float* arow = &hL[mrow][0];
    v8f acc = zero8();
#pragma unroll
    for (int k0=0;k0<128;k0+=32) acc = wmma_bf16(a_frag_f32(arow,k0), b_frag(wtk2,128,n0,k0), acc);
    const float bv = k2b[n];
    const float invs = 0.08838834764831845f; // 1/sqrt(128)
#pragma unroll
    for (int j=0;j<8;++j){
      int m = mh+j;
      float kj = acc[j]+bv;
      float qv = qb[(size_t)dids[m]*128 + n];
      alpha[(size_t)(e0+m)*128 + n] = qv*kj*invs;
    }
  }
  __syncthreads();
  { // h = silu([v[dst], v[src], ee] @ msg1 + b)
    const float* ad = vb + (size_t)dids[mrow]*128;
    const float* as = vb + (size_t)sids[mrow]*128;
    const float* ae = &eeL[mrow][0];
    v8f acc = zero8();
#pragma unroll
    for (int c=0;c<4;++c) acc = wmma_bf16(a_frag_f32(ad,c*32), b_frag(wtm1,384,n0,c*32), acc);
#pragma unroll
    for (int c=0;c<4;++c) acc = wmma_bf16(a_frag_f32(as,c*32), b_frag(wtm1,384,n0,128+c*32), acc);
#pragma unroll
    for (int c=0;c<4;++c) acc = wmma_bf16(a_frag_f32(ae,c*32), b_frag(wtm1,384,n0,256+c*32), acc);
    const float bv = m1b[n];
#pragma unroll
    for (int j=0;j<8;++j) hL[mh+j][n] = silu_f(acc[j]+bv);
  }
  __syncthreads();
  { // msg = h @ msg2 + b
    const float* arow = &hL[mrow][0];
    v8f acc = zero8();
#pragma unroll
    for (int k0=0;k0<128;k0+=32) acc = wmma_bf16(a_frag_f32(arow,k0), b_frag(wtm2,128,n0,k0), acc);
    const float bv = m2b[n];
#pragma unroll
    for (int j=0;j<8;++j) msg[(size_t)(e0+mh+j)*128 + n] = acc[j]+bv;
  }
}

// ---------------- column mean/var partials (128-wide tensors) ----------------
__global__ void colstats_kernel(const float* __restrict__ X, int M, float* __restrict__ stats){
  const int col = (int)threadIdx.x & 127;
  const int rs  = (int)threadIdx.x >> 7;
  float s=0.f, s2=0.f;
  for (int r = (int)blockIdx.x*2 + rs; r < M; r += (int)gridDim.x*2){
    float v = X[(size_t)r*128 + col];
    s += v; s2 += v*v;
  }
  atomicAdd(&stats[col], s);
  atomicAdd(&stats[128+col], s2);
}

// msg * sigmoid(bn(alpha)) scattered into agg[dst]
__global__ void edge_apply_scatter_kernel(const float* __restrict__ alpha, const float* __restrict__ msg,
    const int* __restrict__ dst, const float* __restrict__ stats,
    const float* __restrict__ g, const float* __restrict__ b, float* __restrict__ agg, int E){
  long long i = (long long)blockIdx.x*blockDim.x + threadIdx.x;
  long long tot = (long long)E*128;
  long long s = (long long)gridDim.x*blockDim.x;
  for (; i<tot; i+=s){
    int n = (int)(i & 127);
    int e = (int)(i >> 7);
    float mean = stats[n]/(float)E;
    float var  = stats[128+n]/(float)E - mean*mean;
    float a = (alpha[i]-mean)*rsqrtf(var+1e-5f)*g[n] + b[n];
    atomicAdd(&agg[(size_t)dst[e]*128 + n], msg[i]*sigmoid_f(a));
  }
}

// nf_out = softplus(nf_in + bn(tmp))
__global__ void node_apply_kernel(const float* __restrict__ nfin, const float* __restrict__ tmp,
    const float* __restrict__ stats, const float* __restrict__ g, const float* __restrict__ b,
    float* __restrict__ nfout, int M){
  long long i = (long long)blockIdx.x*blockDim.x + threadIdx.x;
  long long tot = (long long)M*128;
  long long s = (long long)gridDim.x*blockDim.x;
  for (; i<tot; i+=s){
    int n = (int)(i & 127);
    float mean = stats[n]/(float)M;
    float var  = stats[128+n]/(float)M - mean*mean;
    float t = (tmp[i]-mean)*rsqrtf(var+1e-5f)*g[n] + b[n];
    nfout[i] = softplus_f(nfin[i]+t);
  }
}

__global__ void count_kernel(const int* __restrict__ dst, float* __restrict__ cnt, int E){
  int i = blockIdx.x*blockDim.x + threadIdx.x;
  if (i < E) atomicAdd(&cnt[dst[i]], 1.f);
}

__global__ void divcnt_kernel(float* __restrict__ X, const float* __restrict__ cnt, int width, int M){
  long long i = (long long)blockIdx.x*blockDim.x + threadIdx.x;
  long long tot = (long long)M*width;
  long long s = (long long)gridDim.x*blockDim.x;
  for (; i<tot; i+=s){
    int r = (int)(i / width);
    X[i] /= fmaxf(cnt[r],1.f);
  }
}

// ---------------- equivariant pass 1: generate w (16x1536 in LDS), einsum, scatter ----------------
__global__ void equi_pass1_kernel(const float* __restrict__ ef, const float* __restrict__ s,
    const int* __restrict__ src, const int* __restrict__ dst, const float* __restrict__ ea,
    const bf16_t* __restrict__ wt1a, const float* __restrict__ b1a,
    const bf16_t* __restrict__ wt1b, const float* __restrict__ b1b,
    float* __restrict__ f0, float* __restrict__ f1, float* __restrict__ f2){
  extern __shared__ char smem[];
  float (*wl)[1536]  = (float(*)[1536])smem;                       // 98304
  float (*hL)[128]   = (float(*)[128])(smem + 98304);              // 8192
  float (*sjl)[32]   = (float(*)[32])(smem + 98304 + 8192);        // 2048
  float (*sh1s)[3]   = (float(*)[3])(smem + 98304 + 8192 + 2048);  // 192
  float (*sh2s)[5]   = (float(*)[5])(smem + 98304 + 8192 + 2048 + 192); // 320
  int* sids = (int*)(smem + 98304 + 8192 + 2048 + 192 + 320);
  int* dids = sids + 16;
  const int e0 = blockIdx.x*16;
  const int tid = (int)threadIdx.x, wave = tid>>5, lane = tid&31;
  const int mrow = lane&15, mh = (lane>>4)*8;
  if (tid < 16){
    sids[tid] = src[e0+tid]; dids[tid] = dst[e0+tid];
    const float* p = ea + (size_t)(e0+tid)*3;
    float x=p[0], y=p[1], z=p[2];
    float nr = sqrtf(x*x+y*y+z*z)+1e-12f;
    x/=nr; y/=nr; z/=nr;
    const float s3=1.7320508075688772f, s15=3.872983346207417f, s5h=1.118033988749895f;
    sh1s[tid][0]=s3*x; sh1s[tid][1]=s3*y; sh1s[tid][2]=s3*z;
    sh2s[tid][0]=s15*x*y; sh2s[tid][1]=s15*y*z; sh2s[tid][2]=s5h*(3.f*z*z-1.f);
    sh2s[tid][3]=s15*x*z; sh2s[tid][4]=0.5f*s15*(x*x-y*y);
  }
  __syncthreads();
  for (int idx=tid; idx<512; idx+=256){ int e=idx>>5, i=idx&31; sjl[e][i]=s[(size_t)sids[e]*32+i]; }
  { // h = silu(ef @ fc1_a + b)
    const float* arow = ef + (size_t)(e0+mrow)*128;
    int n0 = wave*16, n = n0+mrow;
    v8f acc = zero8();
#pragma unroll
    for (int k0=0;k0<128;k0+=32) acc = wmma_bf16(a_frag_f32(arow,k0), b_frag(wt1a,128,n0,k0), acc);
    float bv = b1a[n];
#pragma unroll
    for (int j=0;j<8;++j) hL[mh+j][n] = silu_f(acc[j]+bv);
  }
  __syncthreads();
  { // w = h @ fc1_b + b  (16x1536 into LDS; 12 tiles per wave)
    const float* arow = &hL[mrow][0];
    for (int t=0;t<12;++t){
      int n0 = (wave*12+t)*16, n = n0+mrow;
      v8f acc = zero8();
#pragma unroll
      for (int k0=0;k0<128;k0+=32) acc = wmma_bf16(a_frag_f32(arow,k0), b_frag(wt1b,128,n0,k0), acc);
      float bv = b1b[n];
#pragma unroll
      for (int j=0;j<8;++j) wl[mh+j][n] = acc[j]+bv;
    }
  }
  __syncthreads();
  const float isns = 0.17677669529663687f; // 1/sqrt(32)
  for (int idx=tid; idx<512; idx+=256){
    int e = idx>>5, o = idx&31;
    float t0=0.f;
    for (int i=0;i<32;++i) t0 += sjl[e][i]*wl[e][i*32+o];
    atomicAdd(&f0[(size_t)dids[e]*32+o], t0*isns);
    if (o < 8){
      float t1=0.f, t2=0.f;
      for (int i=0;i<32;++i){ float sv=sjl[e][i]; t1 += sv*wl[e][1024+i*8+o]; t2 += sv*wl[e][1280+i*8+o]; }
      t1 *= isns; t2 *= isns;
      for (int m=0;m<3;++m) atomicAdd(&f1[(size_t)dids[e]*24+o*3+m], t1*sh1s[e][m]);
      for (int m=0;m<5;++m) atomicAdd(&f2[(size_t)dids[e]*40+o*5+m], t2*sh2s[e][m]);
    }
  }
}

// ---------------- equivariant pass 2 ----------------
__global__ void equi_pass2_kernel(const float* __restrict__ ef,
    const float* __restrict__ f0, const float* __restrict__ f1, const float* __restrict__ f2,
    const int* __restrict__ src, const int* __restrict__ dst, const float* __restrict__ ea,
    const bf16_t* __restrict__ wt2a, const float* __restrict__ b2a,
    const bf16_t* __restrict__ wt2b, const float* __restrict__ b2b,
    float* __restrict__ outacc){
  extern __shared__ char smem[];
  float (*wl)[1536]  = (float(*)[1536])smem;                        // 98304
  float (*hL)[128]   = (float(*)[128])(smem + 98304);               // 8192
  float (*f0s)[32]   = (float(*)[32])(smem + 98304 + 8192);         // 2048
  float (*g1s)[8]    = (float(*)[8])(smem + 98304 + 8192 + 2048);   // 512
  float (*g2s)[8]    = (float(*)[8])(smem + 98304 + 8192 + 2560);   // 512
  float (*sh1s)[3]   = (float(*)[3])(smem + 98304 + 8192 + 3072);   // 192
  float (*sh2s)[5]   = (float(*)[5])(smem + 98304 + 8192 + 3264);   // 320
  int* sids = (int*)(smem + 98304 + 8192 + 3584);
  int* dids = sids + 16;
  const int e0 = blockIdx.x*16;
  const int tid = (int)threadIdx.x, wave = tid>>5, lane = tid&31;
  const int mrow = lane&15, mh = (lane>>4)*8;
  if (tid < 16){
    sids[tid] = src[e0+tid]; dids[tid] = dst[e0+tid];
    const float* p = ea + (size_t)(e0+tid)*3;
    float x=p[0], y=p[1], z=p[2];
    float nr = sqrtf(x*x+y*y+z*z)+1e-12f;
    x/=nr; y/=nr; z/=nr;
    const float s3=1.7320508075688772f, s15=3.872983346207417f, s5h=1.118033988749895f;
    sh1s[tid][0]=s3*x; sh1s[tid][1]=s3*y; sh1s[tid][2]=s3*z;
    sh2s[tid][0]=s15*x*y; sh2s[tid][1]=s15*y*z; sh2s[tid][2]=s5h*(3.f*z*z-1.f);
    sh2s[tid][3]=s15*x*z; sh2s[tid][4]=0.5f*s15*(x*x-y*y);
  }
  __syncthreads();
  { // h = silu(ef @ fc2_a + b)
    const float* arow = ef + (size_t)(e0+mrow)*128;
    int n0 = wave*16, n = n0+mrow;
    v8f acc = zero8();
#pragma unroll
    for (int k0=0;k0<128;k0+=32) acc = wmma_bf16(a_frag_f32(arow,k0), b_frag(wt2a,128,n0,k0), acc);
    float bv = b2a[n];
#pragma unroll
    for (int j=0;j<8;++j) hL[mh+j][n] = silu_f(acc[j]+bv);
  }
  __syncthreads();
  { // w = h @ fc2_b + b
    const float* arow = &hL[mrow][0];
    for (int t=0;t<12;++t){
      int n0 = (wave*12+t)*16, n = n0+mrow;
      v8f acc = zero8();
#pragma unroll
      for (int k0=0;k0<128;k0+=32) acc = wmma_bf16(a_frag_f32(arow,k0), b_frag(wt2b,128,n0,k0), acc);
      float bv = b2b[n];
#pragma unroll
      for (int j=0;j<8;++j) wl[mh+j][n] = acc[j]+bv;
    }
  }
  __syncthreads();
  for (int idx=tid; idx<512; idx+=256){
    int e = idx>>5, i = idx&31;
    f0s[e][i] = f0[(size_t)sids[e]*32+i];
    if (i < 8){
      float a1=0.f, a2=0.f;
      for (int m=0;m<3;++m) a1 += f1[(size_t)sids[e]*24+i*3+m]*sh1s[e][m];
      for (int m=0;m<5;++m) a2 += f2[(size_t)sids[e]*40+i*5+m]*sh2s[e][m];
      g1s[e][i]=a1; g2s[e][i]=a2;
    }
  }
  __syncthreads();
  const float isns=0.17677669529663687f, is24=0.2041241452319315f, is40=0.15811388300841897f;
  for (int idx=tid; idx<512; idx+=256){
    int e = idx>>5, o = idx&31;
    float val=0.f;
    for (int i=0;i<32;++i) val += f0s[e][i]*wl[e][i*32+o];
    val *= isns;
    float v1=0.f, v2=0.f;
    for (int i=0;i<8;++i){ v1 += g1s[e][i]*wl[e][1024+i*32+o]; v2 += g2s[e][i]*wl[e][1280+i*32+o]; }
    val += v1*is24 + v2*is40;
    atomicAdd(&outacc[(size_t)dids[e]*32+o], val);
  }
}

// ---------------- pooling, head, equality adjustment ----------------
__global__ void pool_kernel(const float* __restrict__ nf, const int* __restrict__ batch,
                            float* __restrict__ cf, float* __restrict__ gcnt, int M){
  long long i = (long long)blockIdx.x*blockDim.x + threadIdx.x;
  long long tot = (long long)M*128;
  long long s = (long long)gridDim.x*blockDim.x;
  for (; i<tot; i+=s){
    int n = (int)(i & 127);
    int r = (int)(i >> 7);
    int g = batch[r];
    atomicAdd(&cf[(size_t)g*128+n], nf[i]);
    if (n == 0) atomicAdd(&gcnt[g], 1.f);
  }
}

__global__ void head_kernel(const float* __restrict__ cf, const float* __restrict__ gcnt,
                            const float* __restrict__ fm,
                            const float* __restrict__ w1, const float* __restrict__ b1,
                            const float* __restrict__ w2, const float* __restrict__ b2,
                            float* __restrict__ outp){
  const int g = blockIdx.x;
  const int tid = (int)threadIdx.x;   // 128 threads
  __shared__ float cfl[128], yl[128], hl[128];
  float inv = 1.f / fmaxf(gcnt[g], 1.f);
  cfl[tid] = cf[(size_t)g*128+tid]*inv;
  __syncthreads();
  const float* fmg = fm + (size_t)g*128*128;
  float y = 0.f;
  for (int j=0;j<128;++j) y += fmg[(size_t)tid*128+j]*cfl[j];
  yl[tid] = y;
  __syncthreads();
  float h = b1[tid];
  for (int j=0;j<128;++j) h += yl[j]*w1[(size_t)j*128+tid];
  hl[tid] = silu_f(h);
  __syncthreads();
  if (tid < 36){
    float o = b2[tid];
    for (int j=0;j<128;++j) o += hl[j]*w2[(size_t)j*36+tid];
    outp[(size_t)g*36+tid] = o;
  }
}

__global__ void equality_kernel(const unsigned char* __restrict__ eq,
                                const float* __restrict__ outp, float* __restrict__ d_out){
  int g = blockIdx.x*blockDim.x + threadIdx.x;
  if (g >= NG) return;
  float v[36];
  for (int i=0;i<36;++i) v[i] = outp[(size_t)g*36+i];
  const unsigned char* m1 = eq + (size_t)g*2*36*36;
  const unsigned char* m2 = m1 + 36*36;
  for (int j=0;j<36;++j){
    const unsigned char* row = m1 + j*36;
    float s=0.f, c=0.f;
    for (int k=0;k<36;++k) if (row[k]) { s += v[k]; c += 1.f; }
    float mean = s / fmaxf(c,1.f);
    for (int k=0;k<36;++k) if (row[k]) v[k] = mean;
  }
  for (int j=0;j<36;++j)
    for (int k=0;k<36;++k)
      if (k > j && m2[j*36+k]){
        float a = fabsf(v[j]-v[k])*0.5f;
        float sj = (v[j] < 0.f) ? -a : a;
        v[j] = sj; v[k] = -sj;
      }
  for (int i=0;i<36;++i) d_out[(size_t)g*36+i] = v[i];
}

// ---------------- host orchestration ----------------
extern "C" void kernel_launch(void* const* d_in, const int* in_sizes, int n_in,
                              void* d_out, int out_size, void* d_ws, size_t ws_size,
                              hipStream_t stream){
  (void)in_sizes; (void)n_in; (void)out_size; (void)ws_size;
  auto F = [&](int i){ return (const float*)d_in[i]; };
  // params flattened (jax sorted-key order), then x, edge_index, edge_attr, batch, feat_mask, equality
  const float* x   = F(66);
  const int*   ei  = (const int*)d_in[67];
  const int*   srcp = ei;
  const int*   dstp = ei + NE;
  const float* ea  = F(68);
  const int*   batch = (const int*)d_in[69];
  const float* fm  = F(70);
  const unsigned char* eq = (const unsigned char*)d_in[71];
  float* outfinal = (float*)d_out;

  char* wsbase = (char*)d_ws;
  size_t off = 0;
  auto alloc = [&](size_t bytes)->void*{
    void* p = wsbase + off;
    off += (bytes + 255) & ~(size_t)255;
    return p;
  };
  // bf16 transposed weights
  bf16_t* wt_atom = (bf16_t*)alloc((size_t)128*96*2);
  bf16_t* wt_rbf  = (bf16_t*)alloc((size_t)128*512*2);
  bf16_t *wtq[2], *wtk[2], *wtv[2], *wted[2], *wtcc[2], *wtk1[2], *wtk2[2], *wtm1[2], *wtm2[2];
  for (int c=0;c<2;++c){
    wtq[c]=(bf16_t*)alloc((size_t)128*128*2); wtk[c]=(bf16_t*)alloc((size_t)128*128*2);
    wtv[c]=(bf16_t*)alloc((size_t)128*128*2); wted[c]=(bf16_t*)alloc((size_t)128*128*2);
    wtcc[c]=(bf16_t*)alloc((size_t)128*128*2);
    wtk1[c]=(bf16_t*)alloc((size_t)128*384*2); wtk2[c]=(bf16_t*)alloc((size_t)128*128*2);
    wtm1[c]=(bf16_t*)alloc((size_t)128*384*2); wtm2[c]=(bf16_t*)alloc((size_t)128*128*2);
  }
  bf16_t* wt1a  = (bf16_t*)alloc((size_t)128*128*2);
  bf16_t* wt1b  = (bf16_t*)alloc((size_t)1536*128*2);
  bf16_t* wt2a  = (bf16_t*)alloc((size_t)128*128*2);
  bf16_t* wt2b  = (bf16_t*)alloc((size_t)1536*128*2);
  bf16_t* wtnl  = (bf16_t*)alloc((size_t)32*128*2);
  bf16_t* wtnl2 = (bf16_t*)alloc((size_t)128*32*2);
  bf16_t* wtsk  = (bf16_t*)alloc((size_t)128*128*2);
  // activations (f32)
  float* xp    = (float*)alloc((size_t)NN*96*4);
  float* nfA   = (float*)alloc((size_t)NN*128*4);
  float* nfB   = (float*)alloc((size_t)NN*128*4);
  float* nfF   = (float*)alloc((size_t)NN*128*4);
  float* qb    = (float*)alloc((size_t)NN*128*4);
  float* kb    = (float*)alloc((size_t)NN*128*4);
  float* vb    = (float*)alloc((size_t)NN*128*4);
  float* tmp   = (float*)alloc((size_t)NN*128*4);
  float* agg   = (float*)alloc((size_t)NN*128*4);
  float* ef    = (float*)alloc((size_t)NE*128*4);
  float* alpha = (float*)alloc((size_t)NE*128*4);
  float* msg   = (float*)alloc((size_t)NE*128*4);
  float* sbuf  = (float*)alloc((size_t)NN*32*4);
  float* f0    = (float*)alloc((size_t)NN*32*4);
  float* f1    = (float*)alloc((size_t)NN*24*4);
  float* f2    = (float*)alloc((size_t)NN*40*4);
  float* oacc  = (float*)alloc((size_t)NN*32*4);
  float* cnt   = (float*)alloc((size_t)NN*4);
  float* stats = (float*)alloc(256*4);
  float* cf    = (float*)alloc((size_t)NG*128*4);
  float* gcnt  = (float*)alloc((size_t)NG*4);
  float* hout  = (float*)alloc((size_t)NG*36*4);

  auto prep = [&](int wi, bf16_t* dw, int K, int N, int Kpad){
    int tot = N*Kpad;
    w_to_bf16T_kernel<<<(tot+255)/256, 256, 0, stream>>>(F(wi), dw, K, N, Kpad);
  };
  prep(1,  wt_atom, 92, 128, 96);
  prep(65, wt_rbf, 512, 128, 512);
  for (int c=0;c<2;++c){
    int b = 2 + 22*c;
    prep(b+5,  wtcc[c], 128, 128, 128);
    prep(b+7,  wted[c], 128, 128, 128);
    prep(b+9,  wtk[c],  128, 128, 128);
    prep(b+11, wtk1[c], 384, 128, 384);
    prep(b+13, wtk2[c], 128, 128, 128);
    prep(b+15, wtm1[c], 384, 128, 384);
    prep(b+17, wtm2[c], 128, 128, 128);
    prep(b+19, wtq[c],  128, 128, 128);
    prep(b+21, wtv[c],  128, 128, 128);
  }
  prep(47, wt1a, 128, 128, 128);
  prep(49, wt1b, 128, 1536, 128);
  prep(51, wt2a, 128, 128, 128);
  prep(53, wt2b, 128, 1536, 128);
  prep(55, wtnl, 128, 32, 128);
  prep(57, wtnl2, 32, 128, 32);
  prep(59, wtsk, 128, 128, 128);

  auto zero = [&](float* p, size_t n){ zero_f32_kernel<<<1024,256,0,stream>>>(p,(long long)n); };
  auto gemm = [&](const float* A, int lda, int Kp, const bf16_t* Wt, const float* bias,
                  float* out, int M, int N, int act, int accum){
    dim3 g(M/16, (N+127)/128);
    gemm_kernel<<<g, 256, 0, stream>>>(A, lda, Kp, Wt, bias, out, N, act, accum);
  };

  // 0) pad x to K=96 so the atom-embed GEMM avoids any K-tail guard
  pad_x_kernel<<<2048, 256, 0, stream>>>(x, xp);
  // 1) atom embedding
  gemm(xp, 96, 96, wt_atom, F(0), nfA, NN, 128, 0, 0);
  // 2) RBF + rbf_lin + softplus (fused)
  rbf_ef_kernel<<<NE/16, 256, 0, stream>>>(ea, wt_rbf, F(64), ef);

  // 3) two Comformer conv layers
  auto conv_layer = [&](int c, const float* nin, float* nout){
    int b = 2 + 22*c;
    gemm(nin, 128, 128, wtq[c], F(b+18), qb, NN, 128, 0, 0);
    gemm(nin, 128, 128, wtk[c], F(b+8),  kb, NN, 128, 0, 0);
    gemm(nin, 128, 128, wtv[c], F(b+20), vb, NN, 128, 0, 0);
    conv_edge_kernel<<<NE/16, 256, 0, stream>>>(ef, qb, kb, vb, srcp, dstp,
        wted[c], F(b+6), wtk1[c], F(b+10), wtk2[c], F(b+12), wtm1[c], F(b+14), wtm2[c], F(b+16),
        alpha, msg);
    zero(stats, 256);
    colstats_kernel<<<256, 256, 0, stream>>>(alpha, NE, stats);
    zero(agg, (size_t)NN*128);
    edge_apply_scatter_kernel<<<4096, 256, 0, stream>>>(alpha, msg, dstp, stats, F(b+3), F(b+2), agg, NE);
    gemm(agg, 128, 128, wtcc[c], F(b+4), tmp, NN, 128, 0, 0);
    zero(stats, 256);
    colstats_kernel<<<256, 256, 0, stream>>>(tmp, NN, stats);
    node_apply_kernel<<<2048, 256, 0, stream>>>(nin, tmp, stats, F(b+1), F(b+0), nout, NN);
  };
  conv_layer(0, nfA, nfB);
  conv_layer(1, nfB, nfA);

  // 4) equivariant update
  gemm(nfA, 128, 128, wtnl, F(54), sbuf, NN, 32, 0, 0);
  zero(cnt, NN);
  count_kernel<<<(NE+255)/256, 256, 0, stream>>>(dstp, cnt, NE);
  zero(f0, (size_t)NN*32); zero(f1, (size_t)NN*24); zero(f2, (size_t)NN*40);
  size_t shm1 = 98304 + 8192 + 2048 + 192 + 320 + 128;
  equi_pass1_kernel<<<NE/16, 256, shm1, stream>>>(ef, sbuf, srcp, dstp, ea, wt1a, F(46), wt1b, F(48), f0, f1, f2);
  divcnt_kernel<<<2048, 256, 0, stream>>>(f0, cnt, 32, NN);
  divcnt_kernel<<<2048, 256, 0, stream>>>(f1, cnt, 24, NN);
  divcnt_kernel<<<2048, 256, 0, stream>>>(f2, cnt, 40, NN);
  zero(oacc, (size_t)NN*32);
  size_t shm2 = 98304 + 8192 + 2048 + 512 + 512 + 192 + 320 + 128;
  equi_pass2_kernel<<<NE/16, 256, shm2, stream>>>(ef, f0, f1, f2, srcp, dstp, ea, wt2a, F(50), wt2b, F(52), oacc);
  divcnt_kernel<<<2048, 256, 0, stream>>>(oacc, cnt, 32, NN);
  gemm(nfA, 128, 128, wtsk, F(58), nfF, NN, 128, 0, 0);
  gemm(oacc, 32, 32, wtnl2, F(56), nfF, NN, 128, 0, 1);

  // 5) pooling + head
  zero(cf, (size_t)NG*128); zero(gcnt, NG);
  pool_kernel<<<6400, 256, 0, stream>>>(nfF, batch, cf, gcnt, NN);
  head_kernel<<<NG, 128, 0, stream>>>(cf, gcnt, fm, F(61), F(60), F(63), F(62), hout);

  // 6) equality adjustment
  equality_kernel<<<2, 256, 0, stream>>>(eq, hout, outfinal);
}
